// SparseMoELayer_85289460564192
// MI455X (gfx1250) — matrix operand order repaired and checked by reading, not compiled, not run on hardware
//
#include <hip/hip_runtime.h>
#include <math.h>

#define NT 16384
#define DIM 1024
#define NE 8
#define CAP 2560   // int(1.25 * 16384 / 8)

typedef __bf16 v16bf __attribute__((ext_vector_type(16)));
typedef float  v8f   __attribute__((ext_vector_type(8)));

union FragAB { v16bf v; uint4 q[2]; };

__device__ __forceinline__ unsigned short f2bf(float f) {
    unsigned u = __float_as_uint(f);
    u += 0x7FFFu + ((u >> 16) & 1u);      // round-to-nearest-even
    return (unsigned short)(u >> 16);
}

// ---------------- f32 -> bf16 bulk convert ----------------
__global__ void cvt_f32_bf16(const float* __restrict__ src,
                             unsigned short* __restrict__ dst, int n) {
    int i = (blockIdx.x * blockDim.x + threadIdx.x) * 4;
    if (i >= n) return;
    float4 f = *(const float4*)(src + i);
    ushort4 o;
    o.x = f2bf(f.x); o.y = f2bf(f.y); o.z = f2bf(f.z); o.w = f2bf(f.w);
    *(ushort4*)(dst + i) = o;
}

// ---------------- router: logits, top-2, softmax gates ----------------
__global__ __launch_bounds__(256) void router_kernel(
        const float* __restrict__ x, const float* __restrict__ Wr,
        const float* __restrict__ br,
        int2* __restrict__ t_top, float2* __restrict__ t_gate) {
    __shared__ float WrS[NE * DIM];
    int tid = threadIdx.x;
    for (int i = tid; i < NE * DIM; i += 256) WrS[i] = Wr[i];
    __syncthreads();

    int w = tid >> 5, lane = tid & 31;
    int t = blockIdx.x * 8 + w;
    const float* xr = x + (size_t)t * DIM;

    float xv[32];
#pragma unroll
    for (int i = 0; i < 32; ++i) xv[i] = xr[lane + 32 * i];

    float lg[NE];
#pragma unroll
    for (int e = 0; e < NE; ++e) {
        const float* wp = WrS + e * DIM;
        float p = 0.f;
#pragma unroll
        for (int i = 0; i < 32; ++i) p = fmaf(xv[i], wp[lane + 32 * i], p);
#pragma unroll
        for (int off = 16; off > 0; off >>= 1) p += __shfl_xor(p, off, 32);
        lg[e] = p + br[e];
    }
    // top-2, first index wins ties (matches jax.lax.top_k)
    float v0 = -INFINITY; int i0 = 0;
#pragma unroll
    for (int e = 0; e < NE; ++e) if (lg[e] > v0) { v0 = lg[e]; i0 = e; }
    float v1 = -INFINITY; int i1 = 0;
#pragma unroll
    for (int e = 0; e < NE; ++e) if (e != i0 && lg[e] > v1) { v1 = lg[e]; i1 = e; }
    float ex = __expf(v1 - v0);
    float g0 = 1.f / (1.f + ex);
    float g1 = ex * g0;
    if (lane == 0) {
        t_top[t]  = make_int2(i0, i1);
        t_gate[t] = make_float2(g0, g1);
    }
}

// ---------------- dispatch: stable capacity-limited slot assignment ----------------
__global__ __launch_bounds__(256) void dispatch_kernel(
        const int2* __restrict__ t_top, const float2* __restrict__ t_gate,
        int* __restrict__ order, float* __restrict__ gateS, int* __restrict__ counts) {
    int e = blockIdx.x;
    __shared__ int warp_tot[8];
    __shared__ int running;
    int tid = threadIdx.x, w = tid >> 5, lane = tid & 31;
    if (tid == 0) running = 0;
    __syncthreads();

    for (int base = 0; base < NT; base += 256) {
        int t = base + tid;
        int2 te = t_top[t]; float2 tg = t_gate[t];
        bool f0 = (te.x == e), f1 = (te.y == e);
        bool flag = f0 || f1;
        float g = f0 ? tg.x : (f1 ? tg.y : 0.f);
        unsigned bm = (unsigned)__ballot(flag);
        int pre = __popc(bm & ((1u << lane) - 1u));
        if (lane == 0) warp_tot[w] = __popc(bm);
        __syncthreads();
        int woff = 0;
#pragma unroll
        for (int i = 0; i < 8; ++i) if (i < w) woff += warp_tot[i];
        int pos = running + woff + pre;
        if (flag && pos < CAP) { order[e * CAP + pos] = t; gateS[e * CAP + pos] = g; }
        __syncthreads();
        if (tid == 0) {
            int bt = 0;
            for (int i = 0; i < 8; ++i) bt += warp_tot[i];
            running += bt;
        }
        __syncthreads();
    }
    int cnt = running < CAP ? running : CAP;
    if (tid == 0) counts[e] = cnt;
    for (int s = cnt + tid; s < CAP; s += 256) {
        order[e * CAP + s] = 0; gateS[e * CAP + s] = 0.f;
    }
}

// ---------------- load-balance stats ----------------
__global__ void stats_kernel(const int* __restrict__ counts, float* __restrict__ out_stats) {
    int lane = threadIdx.x;
    float load = (lane < NE) ? (float)counts[lane] : 0.f;
    float s = load;
#pragma unroll
    for (int off = 16; off > 0; off >>= 1) s += __shfl_xor(s, off, 32);
    float dist = load / (s + 1e-8f);
    float term = (lane < NE) ? -dist * logf(dist + 1e-8f) : 0.f;
    float loss = term;
#pragma unroll
    for (int off = 16; off > 0; off >>= 1) loss += __shfl_xor(loss, off, 32);
    if (lane == 0) out_stats[0] = loss;
    if (lane < NE) out_stats[1 + lane] = dist;
}

__global__ void zero_kernel(float4* __restrict__ p, int n4) {
    int i = blockIdx.x * blockDim.x + threadIdx.x;
    if (i < n4) p[i] = make_float4(0.f, 0.f, 0.f, 0.f);
}

// ---------------- expert GEMM: 128x128 tile, 4 waves (2x2), each wave 64x64 = 16 WMMA/k-step ----------------
// LAYER==1: H = relu(gather(x)[order] * W1^T + b1)  -> bf16
// LAYER==2: out[token] += gate * (H * W2^T + b2)    (atomic combine, <=2 adds/elem => deterministic)
template <int LAYER>
__global__ __launch_bounds__(128) void moe_gemm(
        const unsigned short* __restrict__ Abase,
        const unsigned short* __restrict__ Wb,
        const float* __restrict__ bias,
        const int* __restrict__ order,
        const float* __restrict__ gateS,
        const int* __restrict__ counts,
        unsigned short* __restrict__ Hout,
        float* __restrict__ out) {
    int e  = blockIdx.z;
    int m0 = blockIdx.y * 128;
    int n0 = blockIdx.x * 128;
    int cnt = counts[e];
    if (m0 >= cnt) return;   // tile entirely past this expert's load

    __shared__ __align__(16) unsigned short At[128 * 40];  // 128 rows x 32 k (+8 pad)
    __shared__ __align__(16) unsigned short Bt[128 * 40];  // 128 n-rows x 32 k (+8 pad)
    __shared__ int tokS[128];

    int tid = threadIdx.x;
    int w = tid >> 5, lane = tid & 31;
    int wr = w >> 1, wc = w & 1;

    if (LAYER == 1) tokS[tid] = order[e * CAP + m0 + tid];
    __syncthreads();

    // staging: thread t owns row t (32 halfs = 64 B) of both tiles
    const unsigned short* arow =
        (LAYER == 1) ? (Abase + (size_t)tokS[tid] * DIM)
                     : (Abase + ((size_t)e * CAP + m0 + tid) * DIM);
    const unsigned short* brow = Wb + ((size_t)e * DIM + n0 + tid) * DIM;

    v8f acc[4][4];
    v8f zero = {0.f, 0.f, 0.f, 0.f, 0.f, 0.f, 0.f, 0.f};
#pragma unroll
    for (int i = 0; i < 4; ++i)
#pragma unroll
        for (int j = 0; j < 4; ++j) acc[i][j] = zero;

    for (int k0 = 0; k0 < DIM; k0 += 32) {
        uint4 a[4], b[4];
#pragma unroll
        for (int q = 0; q < 4; ++q) {
            a[q] = *(const uint4*)(arow + k0 + q * 8);
            b[q] = *(const uint4*)(brow + k0 + q * 8);
        }
        __syncthreads();                       // previous iter's frag reads done
#pragma unroll
        for (int q = 0; q < 4; ++q) {
            *(uint4*)&At[tid * 40 + q * 8] = a[q];
            *(uint4*)&Bt[tid * 40 + q * 8] = b[q];
        }
        __syncthreads();

        // A 16x32 bf16 frag: lanes 0-15 hold K 0..7/16..23, lanes 16-31 hold K 8..15/24..31
        FragAB fa[4];
        int am  = (lane & 15) + wr * 64;
        int ko8 = (lane >> 4) * 8;
#pragma unroll
        for (int mi = 0; mi < 4; ++mi) {
            int m = am + mi * 16;
            fa[mi].q[0] = *(const uint4*)&At[m * 40 + ko8];
            fa[mi].q[1] = *(const uint4*)&At[m * 40 + 16 + ko8];
        }
        // B 32x16 bf16 frag: lanes 0-15 = N, K 0..7/16..23; lanes 16-31 = N, K 8..15/24..31
        int bn   = (lane & 15) + wc * 64;
        int ko16 = (lane >> 4) * 8;
#pragma unroll
        for (int ni = 0; ni < 4; ++ni) {
            FragAB fb;
            int n = bn + ni * 16;
            fb.q[0] = *(const uint4*)&Bt[n * 40 + ko16];
            fb.q[1] = *(const uint4*)&Bt[n * 40 + ko16 + 16];
#pragma unroll
            for (int mi = 0; mi < 4; ++mi)
                acc[mi][ni] = __builtin_amdgcn_wmma_f32_16x16x32_bf16(
                    false, fa[mi].v, false, fb.v,
                    (short)0, acc[mi][ni], false, false);
        }
    }

    // epilogue — C/D layout: lanes 0-15: N=lane, M=r; lanes 16-31: N=lane-16, M=8+r
#pragma unroll
    for (int mi = 0; mi < 4; ++mi) {
#pragma unroll
        for (int ni = 0; ni < 4; ++ni) {
            int n = n0 + wc * 64 + ni * 16 + (lane & 15);
            float bv = bias[(size_t)e * DIM + n];
#pragma unroll
            for (int r = 0; r < 8; ++r) {
                int ml = (lane < 16) ? r : (r + 8);
                int m = m0 + wr * 64 + mi * 16 + ml;
                float v = acc[mi][ni][r] + bv;
                if (LAYER == 1) {
                    v = fmaxf(v, 0.f);
                    Hout[((size_t)e * CAP + m) * DIM + n] = f2bf(v);
                } else {
                    float g = gateS[e * CAP + m];
                    if (g != 0.f) {
                        int tok = order[e * CAP + m];
                        atomicAdd(out + (size_t)tok * DIM + n, g * v);
                    }
                }
            }
        }
    }
}

extern "C" void kernel_launch(void* const* d_in, const int* in_sizes, int n_in,
                              void* d_out, int out_size, void* d_ws, size_t ws_size,
                              hipStream_t stream) {
    const float* x  = (const float*)d_in[0];
    const float* Wr = (const float*)d_in[1];
    const float* br = (const float*)d_in[2];
    const float* W1 = (const float*)d_in[3];
    const float* b1 = (const float*)d_in[4];
    const float* W2 = (const float*)d_in[5];
    const float* b2 = (const float*)d_in[6];
    float* out = (float*)d_out;

    char* ws = (char*)d_ws;
    unsigned short* xb   = (unsigned short*)(ws);                 // 32 MB
    unsigned short* W1b  = (unsigned short*)(ws + 33554432ull);   // 16 MB
    unsigned short* W2b  = (unsigned short*)(ws + 50331648ull);   // 16 MB
    unsigned short* Hb   = (unsigned short*)(ws + 67108864ull);   // 40 MB  (8*2560*1024 bf16)
    int*    order  = (int*)   (ws + 109051904ull);                // 80 KB
    float*  gateS  = (float*) (ws + 109133824ull);                // 80 KB
    int*    counts = (int*)   (ws + 109215744ull);                // 32 B (padded)
    int2*   t_top  = (int2*)  (ws + 109216000ull);                // 128 KB
    float2* t_gate = (float2*)(ws + 109347072ull);                // 128 KB

    cvt_f32_bf16<<<NT * DIM / 1024, 256, 0, stream>>>(x,  xb,  NT * DIM);
    cvt_f32_bf16<<<NE * DIM * DIM / 1024, 256, 0, stream>>>(W1, W1b, NE * DIM * DIM);
    cvt_f32_bf16<<<NE * DIM * DIM / 1024, 256, 0, stream>>>(W2, W2b, NE * DIM * DIM);

    router_kernel<<<NT / 8, 256, 0, stream>>>(x, Wr, br, t_top, t_gate);
    dispatch_kernel<<<NE, 256, 0, stream>>>(t_top, t_gate, order, gateS, counts);
    stats_kernel<<<1, 32, 0, stream>>>(counts, out + (size_t)NT * DIM);
    zero_kernel<<<NT * DIM / 4 / 256, 256, 0, stream>>>((float4*)out, NT * DIM / 4);

    dim3 g(DIM / 128, CAP / 128, NE);
    moe_gemm<1><<<g, 128, 0, stream>>>(xb, W1b, b1, order, gateS, counts, Hb, out);
    moe_gemm<2><<<g, 128, 0, stream>>>(Hb, W2b, b2, order, gateS, counts, Hb, out);
}
